// MultiHeadedAttention_9440338117174
// MI455X (gfx1250) — matrix operand order repaired and checked by reading, compiled, not run on hardware
//
#include <hip/hip_runtime.h>
#include <cstdint>
#include <cstddef>

// ---------------------------------------------------------------------------
// Problem constants (B=4, S=2048, D=1024, H=8, dk=128)
// ---------------------------------------------------------------------------
#define BB   4
#define SS   2048
#define DD   1024
#define HH   8
#define DK   128
#define BS   (BB * SS)        // 8192 rows
#define DH   (DD * HH)        // 8192 (Wv output width)
#define LN_EPS 1e-5f

typedef __bf16 v16bf __attribute__((ext_vector_type(16)));
typedef float  v8f   __attribute__((ext_vector_type(8)));

union FragB {              // one 16x32 bf16 WMMA operand fragment (8 dwords/lane)
    uint4 q[2];
    v16bf v;
};

__device__ __forceinline__ uint16_t f2bf(float f) {
    uint32_t u = __float_as_uint(f);
    uint32_t r = u + 0x7FFFu + ((u >> 16) & 1u);   // round-to-nearest-even
    return (uint16_t)(r >> 16);
}

__device__ __forceinline__ v8f wmma_bf16(const FragB& a, const FragB& b, v8f c) {
    return __builtin_amdgcn_wmma_f32_16x16x32_bf16(
        false, a.v, false, b.v, (short)0, c, false, false);
}

// ---------------------------------------------------------------------------
// K0: transpose + f32->bf16 convert:  W[K][N] (f32) -> Wt[N][K] (bf16 bits)
// ---------------------------------------------------------------------------
__global__ __launch_bounds__(256)
void transpose_convert_k(const float* __restrict__ src, uint16_t* __restrict__ dst,
                         int K, int N) {
    __shared__ float tile[32][33];
    int n0 = blockIdx.x * 32;
    int k0 = blockIdx.y * 32;
    int tx = threadIdx.x, ty = threadIdx.y;     // (32, 8)
#pragma unroll
    for (int r = 0; r < 4; ++r) {
        int k = k0 + ty + r * 8;
        tile[ty + r * 8][tx] = src[(size_t)k * N + n0 + tx];
    }
    __syncthreads();
#pragma unroll
    for (int r = 0; r < 4; ++r) {
        int n = n0 + ty + r * 8;
        dst[(size_t)n * K + k0 + tx] = f2bf(tile[tx][ty + r * 8]);
    }
}

// ---------------------------------------------------------------------------
// K1: LayerNorm -> bf16.  One 256-thread block per row of D=1024.
// ---------------------------------------------------------------------------
__global__ __launch_bounds__(256)
void layernorm_k(const float* __restrict__ x, const float* __restrict__ gamma,
                 const float* __restrict__ beta, uint16_t* __restrict__ xn) {
    __shared__ float red[256];
    int row = blockIdx.x;
    int tid = threadIdx.x;
    const float* xr = x + (size_t)row * DD;

    float v[4];
    float s = 0.f;
#pragma unroll
    for (int k = 0; k < 4; ++k) { v[k] = xr[tid + k * 256]; s += v[k]; }
    red[tid] = s; __syncthreads();
    for (int off = 128; off > 0; off >>= 1) {
        if (tid < off) red[tid] += red[tid + off];
        __syncthreads();
    }
    float mu = red[0] * (1.0f / DD);
    __syncthreads();

    float s2 = 0.f;
#pragma unroll
    for (int k = 0; k < 4; ++k) { float d = v[k] - mu; s2 += d * d; }
    red[tid] = s2; __syncthreads();
    for (int off = 128; off > 0; off >>= 1) {
        if (tid < off) red[tid] += red[tid + off];
        __syncthreads();
    }
    float rstd = rsqrtf(red[0] * (1.0f / DD) + LN_EPS);

#pragma unroll
    for (int k = 0; k < 4; ++k) {
        int d = tid + k * 256;
        float y = (v[k] - mu) * rstd * gamma[d] + beta[d];
        xn[(size_t)row * DD + d] = f2bf(y);
    }
}

// ---------------------------------------------------------------------------
// K2: bf16 WMMA GEMM:  C = A[M][K] @ Wt[N][K]^T + bias
//     block tile 128x128, 8 waves, each wave 32x64 (2x4 WMMA acc tiles)
//     mode 0: C row-major bf16 [M][N]            (Q, K projections)
//     mode 1: C in V-transposed layout [B][H][D][S] (V projection)
// ---------------------------------------------------------------------------
__global__ __launch_bounds__(256)
void gemm_bf16_k(const uint16_t* __restrict__ A, const uint16_t* __restrict__ Wt,
                 const float* __restrict__ bias, uint16_t* __restrict__ C,
                 int M, int N, int K, int mode) {
    int tid  = threadIdx.x;
    int lane = tid & 31;
    int w    = tid >> 5;
    int wm = w & 3, wn = w >> 2;
    int rowBase = blockIdx.x * 128 + wm * 32;
    int colBase = blockIdx.y * 128 + wn * 64;
    int lm = lane & 15, kh = lane >> 4;

    v8f acc[2][4] = {};

    for (int kk0 = 0; kk0 < K; kk0 += 32) {
        FragB a[2];
#pragma unroll
        for (int i = 0; i < 2; ++i) {
            const uint16_t* p = A + (size_t)(rowBase + i * 16 + lm) * K + kk0 + kh * 8;
            a[i].q[0] = *(const uint4*)(p);
            a[i].q[1] = *(const uint4*)(p + 16);
        }
        FragB bf[4];
#pragma unroll
        for (int j = 0; j < 4; ++j) {
            const uint16_t* p = Wt + (size_t)(colBase + j * 16 + lm) * K + kk0 + kh * 8;
            bf[j].q[0] = *(const uint4*)(p);
            bf[j].q[1] = *(const uint4*)(p + 16);
        }
#pragma unroll
        for (int i = 0; i < 2; ++i)
#pragma unroll
            for (int j = 0; j < 4; ++j)
                acc[i][j] = wmma_bf16(a[i], bf[j], acc[i][j]);
    }

#pragma unroll
    for (int i = 0; i < 2; ++i) {
#pragma unroll
        for (int j = 0; j < 4; ++j) {
            int ng = colBase + j * 16 + lm;
            float bv = bias[ng];
            if (mode == 0) {
#pragma unroll
                for (int r = 0; r < 8; ++r) {
                    int mg = rowBase + i * 16 + kh * 8 + r;
                    C[(size_t)mg * N + ng] = f2bf(acc[i][j][r] + bv);
                }
            } else {
                // Vt[((b*H + h)*D + d) * S + t],  b=mg>>11, t=mg&2047, h=ng>>10, d=ng&1023
                int hb = ng >> 10, d = ng & 1023;
                int mg0 = rowBase + i * 16 + kh * 8;
                int b  = mg0 >> 11, t0 = mg0 & 2047;
                uint32_t pk[4];
#pragma unroll
                for (int r = 0; r < 4; ++r) {
                    uint32_t lo = f2bf(acc[i][j][2 * r] + bv);
                    uint32_t hi = f2bf(acc[i][j][2 * r + 1] + bv);
                    pk[r] = lo | (hi << 16);
                }
                uint16_t* dst = C + (((size_t)(b * HH + hb) * DD + d) << 11) + t0;
                *(uint4*)dst = make_uint4(pk[0], pk[1], pk[2], pk[3]);
            }
        }
    }
}

// ---------------------------------------------------------------------------
// K3: attention for one (b, h, 32-query tile).
//     LDS: 32x2048 f32 scores (256KB, exploits 320KB WGP LDS) + 32 f32 sums.
//     Phase1: scores = Q Kt / sqrt(dk) via WMMA, each wave owns 256 keys,
//             two 16-row groups share each K fragment.
//     Phase2: rowwise max + exp; unnormalized bf16 probs packed IN PLACE over
//             the f32 score rows (ascending pack: read dword t, write dword
//             t/2 <= t; per-wave in-order LDS makes this hazard-free).
//             1/sum deferred to the phase-3 epilogue.
//     Phase3: att = P @ V (Vt layout), wave owns 128 of 1024 D-columns;
//             V fragments reused across both row groups.
// ---------------------------------------------------------------------------
__global__ __launch_bounds__(256)
void attn_k(const uint16_t* __restrict__ Qn, const uint16_t* __restrict__ Kn,
            const uint16_t* __restrict__ Vt, float* __restrict__ att) {
    extern __shared__ __align__(16) char smem[];
    float*    sc     = (float*)smem;                       // [32][2048] f32
    uint16_t* psc16  = (uint16_t*)smem;                    // bf16 alias, row stride 4096
    float*    sumInv = (float*)(smem + 32 * 2048 * 4);     // [32]

    int b = blockIdx.x >> 3, h = blockIdx.x & 7;
    int q0 = blockIdx.y * 32;
    int tid = threadIdx.x, lane = tid & 31, w = tid >> 5;
    int lm = lane & 15, kh = lane >> 4;

    // ---- Phase 1: scores ----
    FragB qf[2][4];
#pragma unroll
    for (int g = 0; g < 2; ++g)
#pragma unroll
        for (int f = 0; f < 4; ++f) {
            const uint16_t* p = Qn + (size_t)(b * SS + q0 + g * 16 + lm) * DD
                                + h * DK + f * 32 + kh * 8;
            qf[g][f].q[0] = *(const uint4*)(p);
            qf[g][f].q[1] = *(const uint4*)(p + 16);
        }
    const float scale = 0.08838834764831845f;   // 1/sqrt(128)
    for (int i = 0; i < 16; ++i) {              // 16 key-tiles of 16 per wave
        int t0 = w * 256 + i * 16;
        v8f s0 = {}, s1 = {};
#pragma unroll
        for (int f = 0; f < 4; ++f) {
            FragB kf;
            const uint16_t* p = Kn + (size_t)(b * SS + t0 + lm) * DD + h * DK + f * 32 + kh * 8;
            kf.q[0] = *(const uint4*)(p);
            kf.q[1] = *(const uint4*)(p + 16);
            s0 = wmma_bf16(qf[0][f], kf, s0);
            s1 = wmma_bf16(qf[1][f], kf, s1);
        }
#pragma unroll
        for (int r = 0; r < 8; ++r) {
            int m = kh * 8 + r;
            sc[m * 2048 + t0 + lm]          = s0[r] * scale;
            sc[(m + 16) * 2048 + t0 + lm]   = s1[r] * scale;
        }
    }
    __syncthreads();

    // ---- Phase 2: softmax (4 rows per wave), unnormalized bf16 in place ----
#pragma unroll
    for (int rr = 0; rr < 4; ++rr) {
        int m = w * 4 + rr;
        float mx = -3.4e38f;
        for (int t = lane; t < 2048; t += 32) mx = fmaxf(mx, sc[m * 2048 + t]);
#pragma unroll
        for (int off = 16; off > 0; off >>= 1) mx = fmaxf(mx, __shfl_xor(mx, off, 32));
        float sum = 0.f;
        for (int t = lane; t < 2048; t += 32) {
            float e = __expf(sc[m * 2048 + t] - mx);
            sum += e;
            psc16[m * 4096 + t] = f2bf(e);     // in-place pack over f32 row
        }
#pragma unroll
        for (int off = 16; off > 0; off >>= 1) sum += __shfl_xor(sum, off, 32);
        if (lane == 0) sumInv[m] = 1.0f / sum;
    }
    __syncthreads();

    // ---- Phase 3: att = P @ V  (wave owns D-columns [w*128, w*128+128)) ----
    int c0 = w * 128;
    v8f acc[2][8] = {};
    for (int kt = 0; kt < 64; ++kt) {
        int kk0 = kt * 32;
        FragB pf[2];
#pragma unroll
        for (int g = 0; g < 2; ++g) {
            const uint16_t* pp = psc16 + (size_t)(g * 16 + lm) * 4096 + kk0 + kh * 8;
            pf[g].q[0] = *(const uint4*)(pp);
            pf[g].q[1] = *(const uint4*)(pp + 16);
        }
#pragma unroll
        for (int j = 0; j < 8; ++j) {
            FragB vf;
            const uint16_t* vp = Vt + (((size_t)(b * HH + h) * DD + c0 + j * 16 + lm) << 11)
                                 + kk0 + kh * 8;
            vf.q[0] = *(const uint4*)(vp);
            vf.q[1] = *(const uint4*)(vp + 16);
            acc[0][j] = wmma_bf16(pf[0], vf, acc[0][j]);
            acc[1][j] = wmma_bf16(pf[1], vf, acc[1][j]);
        }
    }
#pragma unroll
    for (int g = 0; g < 2; ++g)
#pragma unroll
        for (int j = 0; j < 8; ++j) {
            int d = c0 + j * 16 + lm;
#pragma unroll
            for (int r = 0; r < 8; ++r) {
                int m = g * 16 + kh * 8 + r;
                float inv = sumInv[m];
                att[((size_t)(b * HH + h) * SS + q0 + m) * DD + d] = acc[g][j][r] * inv;
            }
        }
}

// ---------------------------------------------------------------------------
// K4: out_flat[j] = sum_{h=0..7} att_flat[8j + h]
//     (the reference's raw .view(S,B,D,H).sum(-1) collapses to this)
// ---------------------------------------------------------------------------
__global__ __launch_bounds__(256)
void headsum_k(const float* __restrict__ att, float* __restrict__ out, int n) {
    int j = blockIdx.x * 256 + threadIdx.x;
    if (j < n) {
        const float4* p = (const float4*)(att + (size_t)j * 8);
        float4 a = p[0], c = p[1];
        out[j] = a.x + a.y + a.z + a.w + c.x + c.y + c.z + c.w;
    }
}

// ---------------------------------------------------------------------------
// Launch
// ---------------------------------------------------------------------------
extern "C" void kernel_launch(void* const* d_in, const int* in_sizes, int n_in,
                              void* d_out, int out_size, void* d_ws, size_t ws_size,
                              hipStream_t stream) {
    (void)in_sizes; (void)n_in; (void)out_size; (void)ws_size;

    const float* x     = (const float*)d_in[0];
    const float* Wq    = (const float*)d_in[1];
    const float* bq    = (const float*)d_in[2];
    const float* Wk    = (const float*)d_in[3];
    const float* bk    = (const float*)d_in[4];
    const float* Wv    = (const float*)d_in[5];
    const float* bv    = (const float*)d_in[6];
    const float* gamma = (const float*)d_in[7];
    const float* beta  = (const float*)d_in[8];
    float* out = (float*)d_out;

    // Workspace layout (bytes):
    char* ws = (char*)d_ws;
    size_t o = 0;
    uint16_t* xn   = (uint16_t*)(ws + o); o += (size_t)BS * DD * 2;        //  16 MB
    uint16_t* wq_t = (uint16_t*)(ws + o); o += (size_t)DD * DD * 2;        //   2 MB
    uint16_t* wk_t = (uint16_t*)(ws + o); o += (size_t)DD * DD * 2;        //   2 MB
    uint16_t* wv_t = (uint16_t*)(ws + o); o += (size_t)DH * DD * 2;        //  16 MB
    uint16_t* Qn   = (uint16_t*)(ws + o); o += (size_t)BS * DD * 2;        //  16 MB
    uint16_t* Kn   = (uint16_t*)(ws + o); o += (size_t)BS * DD * 2;        //  16 MB
    uint16_t* Vt   = (uint16_t*)(ws + o); o += (size_t)BB * HH * DD * SS * 2; // 128 MB
    float*    att  = (float*)   (ws + o); o += (size_t)BB * HH * SS * DD * 4; // 256 MB

    dim3 tb(32, 8);
    transpose_convert_k<<<dim3(DD / 32, DD / 32), tb, 0, stream>>>(Wq, wq_t, DD, DD);
    transpose_convert_k<<<dim3(DD / 32, DD / 32), tb, 0, stream>>>(Wk, wk_t, DD, DD);
    transpose_convert_k<<<dim3(DH / 32, DD / 32), tb, 0, stream>>>(Wv, wv_t, DD, DH);

    layernorm_k<<<BS, 256, 0, stream>>>(x, gamma, beta, xn);

    gemm_bf16_k<<<dim3(BS / 128, DD / 128), 256, 0, stream>>>(xn, wq_t, bq, Qn, BS, DD, DD, 0);
    gemm_bf16_k<<<dim3(BS / 128, DD / 128), 256, 0, stream>>>(xn, wk_t, bk, Kn, BS, DD, DD, 0);
    gemm_bf16_k<<<dim3(BS / 128, DH / 128), 256, 0, stream>>>(xn, wv_t, bv, Vt, BS, DH, DD, 1);

    // 32x2048 f32 scores + 32 f32 inv-sums = 262272 bytes of dynamic LDS
    attn_k<<<dim3(BB * HH, SS / 32), 256, 32 * 2048 * 4 + 128, stream>>>(Qn, Kn, Vt, att);

    int n = BB * SS * DD;   // 8388608
    headsum_k<<<(n + 255) / 256, 256, 0, stream>>>(att, out, n);
}
